// ImMatchNet_23545010717395
// MI455X (gfx1250) — matrix operand order repaired and checked by reading, not compiled, not run on hardware
//
#include <hip/hip_runtime.h>
#include <hip/hip_bf16.h>

typedef __attribute__((ext_vector_type(16))) _Float16 v16h;
typedef __attribute__((ext_vector_type(8)))  _Float16 v8h;
typedef __attribute__((ext_vector_type(8)))  float    v8f;

#define PM   1024            // positions per batch (32*32) == channels
#define PB   (1u << 20)      // PM*PM elements per batch matrix
#define NB   4

union V16 { v16h v; v8h h[2]; };

// ---------------------------------------------------------------------------
// 1) per-position 1/sqrt(sum_c f^2 + 1e-6).  threads = positions (coalesced).
// ---------------------------------------------------------------------------
__global__ __launch_bounds__(256) void sumsq_kernel(const float* __restrict__ f,
                                                    float* __restrict__ rn) {
  int idx = blockIdx.x * 256 + threadIdx.x;    // over NB*PM
  int b = idx >> 10, p = idx & 1023;
  const float* base = f + ((size_t)b << 20) + p;
  float s = 0.f;
  for (int c = 0; c < PM; ++c) {
    float v = base[(size_t)c << 10];
    s = fmaf(v, v, s);
  }
  rn[idx] = rsqrtf(s + 1e-6f);
}

// ---------------------------------------------------------------------------
// 2) normalize + transpose [b][c][p] f32 -> [b][p][c] f16  (LDS-tiled)
// ---------------------------------------------------------------------------
__global__ __launch_bounds__(256) void norm_t_kernel(const float* __restrict__ f,
                                                     const float* __restrict__ rn,
                                                     _Float16* __restrict__ ft) {
  __shared__ float tile[32][33];
  int b = blockIdx.z;
  int c0 = blockIdx.x * 32, p0 = blockIdx.y * 32;
  int tx = threadIdx.x, ty = threadIdx.y;     // 32 x 8
  const float* src = f + ((size_t)b << 20);
  for (int r = 0; r < 32; r += 8)
    tile[ty + r][tx] = src[(size_t)(c0 + ty + r) * PM + p0 + tx];
  __syncthreads();
  _Float16* dst = ft + ((size_t)b << 20);
  for (int r = 0; r < 32; r += 8) {
    int p = p0 + ty + r, c = c0 + tx;
    dst[(size_t)p * PM + c] = (_Float16)(tile[tx][ty + r] * rn[(b << 10) + p]);
  }
}

// ---------------------------------------------------------------------------
// 3) correlation GEMM: C[b][m][n] = sum_k At[b][m][k] * Bt[b][n][k]
//    One wave -> one 16x16 tile, v_wmma_f32_16x16x32_f16, K loop of 32.
// ---------------------------------------------------------------------------
__global__ __launch_bounds__(256) void corr_wmma_kernel(const _Float16* __restrict__ At,
                                                        const _Float16* __restrict__ Bt,
                                                        float* __restrict__ C) {
  const int lane = threadIdx.x & 31;
  const int wave = threadIdx.x >> 5;
  const int b  = blockIdx.z;
  const int n0 = blockIdx.x * 16;
  const int m0 = (blockIdx.y * 8 + wave) * 16;
  const int half = lane >> 4;      // 0: lanes 0-15, 1: lanes 16-31
  const int l16  = lane & 15;

  const _Float16* arow = At + ((size_t)b << 20) + (size_t)(m0 + l16) * PM;
  const _Float16* brow = Bt + ((size_t)b << 20) + (size_t)(n0 + l16) * PM;

  v8f acc = {};
  for (int kk = 0; kk < PM; kk += 32) {
    V16 a, bm;
    a.h[0] = *(const v8h*)(arow + kk + (half ? 8  : 0));
    a.h[1] = *(const v8h*)(arow + kk + (half ? 24 : 16));
    bm.h[0] = *(const v8h*)(brow + kk + (half ? 8  : 0));
    bm.h[1] = *(const v8h*)(brow + kk + (half ? 24 : 16));
    acc = __builtin_amdgcn_wmma_f32_16x16x32_f16(false, a.v, false, bm.v,
                                                 (short)0, acc, false, false);
  }
  float* crow = C + ((size_t)b << 20) + (size_t)(m0 + half * 8) * PM + n0 + l16;
#pragma unroll
  for (int r = 0; r < 8; ++r) crow[(size_t)r * PM] = acc[r];
}

// ---------------------------------------------------------------------------
// 4) mutual matching: row maxes, col maxes, elementwise apply
// ---------------------------------------------------------------------------
__global__ __launch_bounds__(256) void rowmax_kernel(const float* __restrict__ x,
                                                     float* __restrict__ rmax) {
  int row = blockIdx.x;                           // b*1024 + m
  const float* r = x + (size_t)row * PM;
  float m = -3.4e38f;
  for (int j = threadIdx.x; j < PM; j += 256) m = fmaxf(m, r[j]);
  __shared__ float s[256];
  s[threadIdx.x] = m; __syncthreads();
  for (int off = 128; off > 0; off >>= 1) {
    if (threadIdx.x < off) s[threadIdx.x] = fmaxf(s[threadIdx.x], s[threadIdx.x + off]);
    __syncthreads();
  }
  if (threadIdx.x == 0) rmax[row] = s[0];
}

__global__ __launch_bounds__(256) void colmax_kernel(const float* __restrict__ x,
                                                     float* __restrict__ cmax) {
  int b = blockIdx.y;
  int col = blockIdx.x * 256 + threadIdx.x;
  const float* base = x + ((size_t)b << 20) + col;
  float m = -3.4e38f;
  for (int i = 0; i < PM; ++i) m = fmaxf(m, base[(size_t)i << 10]);
  cmax[(b << 10) + col] = m;
}

__global__ __launch_bounds__(256) void mm_apply_kernel(const float* __restrict__ x,
                                                       const float* __restrict__ rmax,
                                                       const float* __restrict__ cmax,
                                                       float* __restrict__ y) {
  size_t idx = (size_t)blockIdx.x * 256 + threadIdx.x;
  int b = (int)(idx >> 20);
  int m = (int)(idx >> 10) & 1023;
  int n = (int)idx & 1023;
  float v  = x[idx];
  float ra = rmax[(b << 10) + m];
  float rb = cmax[(b << 10) + n];
  y[idx] = v * (v / (ra + 1e-5f)) * (v / (rb + 1e-5f));
}

// ---------------------------------------------------------------------------
// 5) tiled 1024x1024 transpose (per batch) and transpose-add
// ---------------------------------------------------------------------------
__global__ __launch_bounds__(256) void transpose_kernel(const float* __restrict__ x,
                                                        float* __restrict__ xt) {
  __shared__ float tile[32][33];
  int b = blockIdx.z;
  int m0 = blockIdx.y * 32, n0 = blockIdx.x * 32;
  int tx = threadIdx.x, ty = threadIdx.y;
  const float* xb = x + ((size_t)b << 20);
  for (int r = 0; r < 32; r += 8)
    tile[ty + r][tx] = xb[(size_t)(m0 + ty + r) * PM + n0 + tx];
  __syncthreads();
  float* xo = xt + ((size_t)b << 20);
  for (int r = 0; r < 32; r += 8)
    xo[(size_t)(n0 + ty + r) * PM + m0 + tx] = tile[tx][ty + r];
}

__global__ __launch_bounds__(256) void transpose_add_kernel(const float* __restrict__ za,
                                                            const float* __restrict__ zb,
                                                            float* __restrict__ s) {
  __shared__ float tile[32][33];
  int b = blockIdx.z;
  int i0 = blockIdx.y * 32, j0 = blockIdx.x * 32;
  int tx = threadIdx.x, ty = threadIdx.y;
  const float* zbb = zb + ((size_t)b << 20);
  for (int r = 0; r < 32; r += 8)
    tile[ty + r][tx] = zbb[(size_t)(j0 + ty + r) * PM + i0 + tx];
  __syncthreads();
  const float* zab = za + ((size_t)b << 20);
  float* sb = s + ((size_t)b << 20);
  for (int r = 0; r < 32; r += 8) {
    int i = i0 + ty + r, j = j0 + tx;
    sb[(size_t)i * PM + j] = zab[(size_t)i * PM + j] + tile[tx][ty + r];
  }
}

// ---------------------------------------------------------------------------
// 6) direct 4-D conv (3^4 taps, SAME) + ReLU, VALU path (used for conv1/conv3
//    where WMMA padding ratios are hopeless: K=3/32 resp. N=1/16).
// ---------------------------------------------------------------------------
template <int CIN, int COUT>
__global__ __launch_bounds__(256) void conv4d_relu_kernel(const float* __restrict__ x,
                                                          const float* __restrict__ w,
                                                          const float* __restrict__ bias,
                                                          float* __restrict__ y) {
  __shared__ float wl[COUT * CIN * 81];
  for (int idx = threadIdx.x; idx < COUT * CIN * 81; idx += 256) wl[idx] = w[idx];
  __syncthreads();

  int p = blockIdx.x * 256 + threadIdx.x;       // 0 .. 2^20-1
  int l = p & 31, k = (p >> 5) & 31, j = (p >> 10) & 31, i = (p >> 15) & 31;

  float acc[COUT];
#pragma unroll
  for (int co = 0; co < COUT; ++co) acc[co] = bias[co];

  for (int ci = 0; ci < CIN; ++ci) {
    const float* xb = x + ((size_t)ci << 20);
    const float* wc = wl + ci * 81;
    int t = 0;
#pragma unroll
    for (int di = -1; di <= 1; ++di)
#pragma unroll
      for (int dj = -1; dj <= 1; ++dj)
#pragma unroll
        for (int dk = -1; dk <= 1; ++dk)
#pragma unroll
          for (int dl = -1; dl <= 1; ++dl, ++t) {
            unsigned ii = (unsigned)(i + di), jj = (unsigned)(j + dj);
            unsigned kk = (unsigned)(k + dk), ll = (unsigned)(l + dl);
            float xv = 0.f;
            if ((ii < 32u) & (jj < 32u) & (kk < 32u) & (ll < 32u))
              xv = xb[(((ii * 32 + jj) * 32 + kk) * 32) + ll];
#pragma unroll
            for (int co = 0; co < COUT; ++co)
              acc[co] = fmaf(xv, wc[co * CIN * 81 + t], acc[co]);
          }
  }
#pragma unroll
  for (int co = 0; co < COUT; ++co)
    y[((size_t)co << 20) + p] = fmaxf(acc[co], 0.f);
}

// ---------------------------------------------------------------------------
// 7) conv2 via WMMA.
//    Step 1: pack y1 (f32 [10][2^20]) into xcol f16 [2^20][32]:
//            xcol[p][ci*3 + dl+1] = y1[ci][p+dl] (0 at l edges, k=30,31 pad).
//            This folds (ci, dl) into a K=30(pad 32) GEMM contraction so the
//            remaining 27 (di,dj,dk) taps are plain WMMA accumulations with
//            contiguous 64B A-rows.
// ---------------------------------------------------------------------------
__global__ __launch_bounds__(256) void pack_xcol_kernel(const float* __restrict__ y1,
                                                        _Float16* __restrict__ xcol) {
  int p = blockIdx.x * 256 + threadIdx.x;       // 0 .. 2^20-1
  int l = p & 31;
  _Float16 row[32];
#pragma unroll
  for (int q = 0; q < 32; ++q) row[q] = (_Float16)0.f;
#pragma unroll
  for (int ci = 0; ci < 10; ++ci) {
    const float* yb = y1 + ((size_t)ci << 20) + p;
#pragma unroll
    for (int dl = -1; dl <= 1; ++dl) {
      float v = 0.f;
      if ((unsigned)(l + dl) < 32u) v = yb[dl];
      row[ci * 3 + dl + 1] = (_Float16)v;
    }
  }
  _Float16* dst = xcol + ((size_t)p << 5);
  *(v8h*)(dst)      = *(v8h*)(row);
  *(v8h*)(dst + 8)  = *(v8h*)(row + 8);
  *(v8h*)(dst + 16) = *(v8h*)(row + 16);
  *(v8h*)(dst + 24) = *(v8h*)(row + 24);
}

//    Step 2: one wave = 16 positions (along l) x 16 out-channels (10 used).
//    27 uniform-guarded taps, each one v_wmma_f32_16x16x32_f16 with
//    A rows from xcol (2x global b128/lane) and B fragments from LDS-
//    reordered weights W'[tap][co][k] (2x ds b128/lane).
__global__ __launch_bounds__(256) void conv2_wmma_kernel(const _Float16* __restrict__ xcol,
                                                         const float* __restrict__ w,   // [10][10][81]
                                                         const float* __restrict__ bias,
                                                         float* __restrict__ y) {       // [10][2^20]
  __shared__ _Float16 wl[27][16][32];           // 27.6 KB
  for (int idx = threadIdx.x; idx < 27 * 16 * 32; idx += 256) {
    int t = idx >> 9;
    int n = (idx >> 5) & 15;                    // co
    int q = idx & 31;                           // ci*3 + dl+1
    float v = 0.f;
    if (n < 10 && q < 30) {
      int ci = q / 3, dl = q - ci * 3;          // dl in 0..2
      v = w[(n * 10 + ci) * 81 + t * 3 + dl];
    }
    wl[t][n][q] = (_Float16)v;
  }
  __syncthreads();

  const int lane = threadIdx.x & 31;
  const int wave = threadIdx.x >> 5;
  const int half = lane >> 4;
  const int l16  = lane & 15;

  int tile = blockIdx.x * 8 + wave;             // 0 .. 65535
  int p0 = tile << 4;
  int i = p0 >> 15, j = (p0 >> 10) & 31, kz = (p0 >> 5) & 31;

  float bz = (l16 < 10) ? bias[l16] : 0.f;
  v8f acc;
#pragma unroll
  for (int r = 0; r < 8; ++r) acc[r] = bz;

  const _Float16* arow0 = xcol + ((size_t)(p0 + l16) << 5);

#pragma unroll
  for (int di = -1; di <= 1; ++di) {
    if ((unsigned)(i + di) >= 32u) continue;    // wave-uniform: EXEC stays full
#pragma unroll
    for (int dj = -1; dj <= 1; ++dj) {
      if ((unsigned)(j + dj) >= 32u) continue;
#pragma unroll
      for (int dk = -1; dk <= 1; ++dk) {
        if ((unsigned)(kz + dk) >= 32u) continue;
        const int t = (di + 1) * 9 + (dj + 1) * 3 + (dk + 1);
        const ptrdiff_t off = ((ptrdiff_t)(di * 32768 + dj * 1024 + dk * 32)) * 32;
        const _Float16* ar = arow0 + off;
        V16 a, bm;
        a.h[0] = *(const v8h*)(ar + (half ? 8  : 0));
        a.h[1] = *(const v8h*)(ar + (half ? 24 : 16));
        const _Float16* br = &wl[t][l16][0];
        bm.h[0] = *(const v8h*)(br + (half ? 8  : 0));
        bm.h[1] = *(const v8h*)(br + (half ? 24 : 16));
        acc = __builtin_amdgcn_wmma_f32_16x16x32_f16(false, a.v, false, bm.v,
                                                     (short)0, acc, false, false);
      }
    }
  }

  if (l16 < 10) {                               // co = l16; positions contiguous
    float* yp = y + ((size_t)l16 << 20) + p0 + half * 8;
#pragma unroll
    for (int r = 0; r < 8; ++r) yp[r] = fmaxf(acc[r], 0.f);
  }
}

// ---------------------------------------------------------------------------
// host orchestration
// ---------------------------------------------------------------------------
static void run_mutual_matching(const float* x, float* y, float* rmax, float* cmax,
                                hipStream_t s) {
  rowmax_kernel<<<NB * PM, 256, 0, s>>>(x, rmax);
  colmax_kernel<<<dim3(PM / 256, NB), 256, 0, s>>>(x, cmax);
  mm_apply_kernel<<<(NB * PB) / 256, 256, 0, s>>>(x, rmax, cmax, y);
}

extern "C" void kernel_launch(void* const* d_in, const int* in_sizes, int n_in,
                              void* d_out, int out_size, void* d_ws, size_t ws_size,
                              hipStream_t stream) {
  const float* fA = (const float*)d_in[0];
  const float* fB = (const float*)d_in[1];
  const float* W1 = (const float*)d_in[2];
  const float* B1 = (const float*)d_in[3];
  const float* W2 = (const float*)d_in[4];
  const float* B2 = (const float*)d_in[5];
  const float* W3 = (const float*)d_in[6];
  const float* B3 = (const float*)d_in[7];
  float* out = (float*)d_out;

  char* ws = (char*)d_ws;
  size_t off = 0;
  auto take = [&](size_t bytes) -> char* {
    char* p = ws + off;
    off = (off + bytes + 255) & ~(size_t)255;
    return p;
  };

  _Float16* At   = (_Float16*)take((size_t)NB * PB * 2);   // 8 MB
  _Float16* Bt   = (_Float16*)take((size_t)NB * PB * 2);   // 8 MB
  float* rnA     = (float*)take((size_t)NB * PM * 4);
  float* rnB     = (float*)take((size_t)NB * PM * 4);
  float* X       = (float*)take((size_t)NB * PB * 4);      // corr -> mm in place; later S
  float* XT      = (float*)take((size_t)NB * PB * 4);
  float* RMAX    = (float*)take((size_t)NB * PM * 4);
  float* CMAX    = (float*)take((size_t)NB * PM * 4);
  float* Y1      = (float*)take((size_t)10 * PB * 4);      // 42 MB
  float* Y2      = (float*)take((size_t)10 * PB * 4);      // 42 MB
  _Float16* XCOL = (_Float16*)take((size_t)PB * 32 * 2);   // 64 MB
  float* ZA      = (float*)take((size_t)NB * PB * 4);
  float* ZB      = (float*)take((size_t)NB * PB * 4);
  (void)ws_size; (void)in_sizes; (void)n_in; (void)out_size;

  // 1-2) L2 normalize + transpose to f16 pos-major
  sumsq_kernel<<<(NB * PM) / 256, 256, 0, stream>>>(fA, rnA);
  sumsq_kernel<<<(NB * PM) / 256, 256, 0, stream>>>(fB, rnB);
  norm_t_kernel<<<dim3(32, 32, NB), dim3(32, 8), 0, stream>>>(fA, rnA, At);
  norm_t_kernel<<<dim3(32, 32, NB), dim3(32, 8), 0, stream>>>(fB, rnB, Bt);

  // 3) correlation via WMMA
  corr_wmma_kernel<<<dim3(64, 8, NB), 256, 0, stream>>>(At, Bt, X);

  // 4) first mutual matching (in place)
  run_mutual_matching(X, X, RMAX, CMAX, stream);

  // 5) transposed copy for symmetric branch
  transpose_kernel<<<dim3(32, 32, NB), dim3(32, 8), 0, stream>>>(X, XT);

  // 6) neighbourhood-consensus stacks, per batch / per branch
  for (int b = 0; b < NB; ++b) {
    for (int br = 0; br < 2; ++br) {
      const float* xin = (br ? XT : X) + ((size_t)b << 20);
      float* zout = (br ? ZB : ZA) + ((size_t)b << 20);
      conv4d_relu_kernel<1, 10><<<PB / 256, 256, 0, stream>>>(xin, W1, B1, Y1);
      pack_xcol_kernel<<<PB / 256, 256, 0, stream>>>(Y1, XCOL);
      conv2_wmma_kernel<<<(PB / 16) / 8, 256, 0, stream>>>(XCOL, W2, B2, Y2);
      conv4d_relu_kernel<10, 1><<<PB / 256, 256, 0, stream>>>(Y2, W3, B3, zout);
    }
  }

  // 7) symmetric combine: S = ZA + ZB^T  (reuse X; conv inputs no longer needed)
  float* S = X;
  transpose_add_kernel<<<dim3(32, 32, NB), dim3(32, 8), 0, stream>>>(ZA, ZB, S);

  // 8) final mutual matching -> d_out
  run_mutual_matching(S, out, RMAX, CMAX, stream);
}
